// GAT_20598663152070
// MI455X (gfx1250) — matrix operand order repaired and checked by reading, not compiled, not run on hardware
//
#include <hip/hip_runtime.h>

typedef __bf16 bf16_t;
typedef __attribute__((ext_vector_type(16))) __bf16 v16bf;
typedef __attribute__((ext_vector_type(8)))  __bf16 v8bf;
typedef __attribute__((ext_vector_type(8)))  float   v8f;

#define NEG_SLOPE 0.2f

// ---- monotone float <-> uint encoding so atomicMax(u32) == float max ----
__device__ __forceinline__ unsigned enc_f32(float f) {
  unsigned u = __float_as_uint(f);
  return u ^ ((u >> 31) ? 0xFFFFFFFFu : 0x80000000u);
}
__device__ __forceinline__ float dec_f32(unsigned u) {
  u ^= ((u >> 31) ? 0x80000000u : 0xFFFFFFFFu);
  return __uint_as_float(u);
}
// enc(-INFINITY) == 0x007FFFFF (host constant used for init)

__device__ __forceinline__ v16bf frag2(const bf16_t* p0, const bf16_t* p1) {
  v8bf lo = *(const v8bf*)p0;
  v8bf hi = *(const v8bf*)p1;
  return __builtin_shufflevector(lo, hi, 0,1,2,3,4,5,6,7,8,9,10,11,12,13,14,15);
}

__global__ void k_fill_u32(unsigned* p, unsigned v, long long n) {
  long long i = (long long)blockIdx.x * blockDim.x + threadIdx.x;
  if (i < n) p[i] = v;
}

__global__ void k_f32_to_bf16(const float* __restrict__ in, bf16_t* __restrict__ out,
                              long long n) {
  long long i = (long long)blockIdx.x * blockDim.x + threadIdx.x;
  if (i < n) out[i] = (bf16_t)in[i];
}

// W [K, Nout] f32 (row-major) -> Wt [Nout, K] bf16 (row-major)
__global__ void k_transpose_w(const float* __restrict__ W, bf16_t* __restrict__ Wt,
                              int K, int Nout) {
  long long i = (long long)blockIdx.x * blockDim.x + threadIdx.x;
  long long total = (long long)K * Nout;
  if (i >= total) return;
  int k = (int)(i / Nout);
  int n = (int)(i % Nout);
  Wt[(long long)n * K + k] = (bf16_t)W[i];
}

// C[M,Nout] = A[M,K] * Bt[Nout,K]^T using v_wmma_f32_16x16x32_bf16.
// One wave computes a 16x64 strip (4 N-tiles): A fragment loaded once per
// k-step and reused by 4 WMMAs. Requires Nout % 64 == 0 (Nout = H*C = 1024).
__global__ void k_gemm_wmma(const bf16_t* __restrict__ A, const bf16_t* __restrict__ Bt,
                            float* __restrict__ C, int M, int K, int Nout) {
  int wave = (int)((blockIdx.x * (long long)blockDim.x + threadIdx.x) >> 5);
  int lane = threadIdx.x & 31;
  int ngrp   = Nout >> 6;          // groups of 4 N-tiles
  int tilesM = (M + 15) >> 4;
  if (wave >= tilesM * ngrp) return;          // uniform per wave
  int tM = (wave / ngrp) << 4;
  int tN = (wave % ngrp) << 6;
  int lm = lane & 15;
  int hi = lane >> 4;

  int arowi = tM + lm; if (arowi >= M) arowi = M - 1;   // clamp (M%16==0 here)
  // A frag (16x32 bf16): lane holds row lm; elems hold K = hi*16 + 0..15 (contiguous)
  const bf16_t* arow = A + (long long)arowi * K + hi * 16;
  // B frag (32x16 bf16): lane holds col lm; elems hold K = hi*16 + 0..15 (contiguous)
  const bf16_t* b0p = Bt + (long long)(tN +  0 + lm) * K + hi * 16;
  const bf16_t* b1p = Bt + (long long)(tN + 16 + lm) * K + hi * 16;
  const bf16_t* b2p = Bt + (long long)(tN + 32 + lm) * K + hi * 16;
  const bf16_t* b3p = Bt + (long long)(tN + 48 + lm) * K + hi * 16;

  v8f c0 = {}, c1 = {}, c2 = {}, c3 = {};
  for (int k0 = 0; k0 < K; k0 += 32) {
    v16bf a = frag2(arow + k0, arow + k0 + 8);
    v16bf b;
    b = frag2(b0p + k0, b0p + k0 + 8);
    c0 = __builtin_amdgcn_wmma_f32_16x16x32_bf16(false, a, false, b, (short)0, c0, false, false);
    b = frag2(b1p + k0, b1p + k0 + 8);
    c1 = __builtin_amdgcn_wmma_f32_16x16x32_bf16(false, a, false, b, (short)0, c1, false, false);
    b = frag2(b2p + k0, b2p + k0 + 8);
    c2 = __builtin_amdgcn_wmma_f32_16x16x32_bf16(false, a, false, b, (short)0, c2, false, false);
    b = frag2(b3p + k0, b3p + k0 + 8);
    c3 = __builtin_amdgcn_wmma_f32_16x16x32_bf16(false, a, false, b, (short)0, c3, false, false);
  }

  // C/D layout: vgpr r, lanes 0-15 -> (M=r, N=lm); lanes 16-31 -> (M=r+8, N=lm)
  float* crow = C + (long long)tM * Nout + tN;
  #pragma unroll
  for (int r = 0; r < 8; ++r) {
    int gr = tM + r + hi * 8;
    if (gr < M) {
      long long ro = (long long)(r + hi * 8) * Nout + lm;
      crow[ro]      = c0[r];
      crow[ro + 16] = c1[r];
      crow[ro + 32] = c2[r];
      crow[ro + 48] = c3[r];
    }
  }
}

// e_src[n,h] = sum_c h[n,h,c]*a_src[h,c] ; e_dst likewise. One wave per (n,h).
__global__ void k_node_logits(const float* __restrict__ h, const float* __restrict__ a_src,
                              const float* __restrict__ a_dst, float* __restrict__ es,
                              float* __restrict__ ed, int Nn, int Hh, int Cc) {
  int wave = (int)((blockIdx.x * (long long)blockDim.x + threadIdx.x) >> 5);
  int lane = threadIdx.x & 31;
  if (wave >= Nn * Hh) return;
  int n = wave / Hh, hh = wave % Hh;
  const float* hp  = h + ((long long)n * Hh + hh) * Cc;
  const float* asp = a_src + (long long)hh * Cc;
  const float* adp = a_dst + (long long)hh * Cc;
  float s0 = 0.f, s1 = 0.f;
  for (int c = lane; c < Cc; c += 32) {
    float v = hp[c];
    s0 += v * asp[c];
    s1 += v * adp[c];
  }
  #pragma unroll
  for (int off = 16; off > 0; off >>= 1) {
    s0 += __shfl_xor(s0, off, 32);
    s1 += __shfl_xor(s1, off, 32);
  }
  if (lane == 0) { es[wave] = s0; ed[wave] = s1; }
}

// pass 1: segment max of leaky_relu(e_src[src]+e_dst[dst]) into menc[dst,h]
__global__ void k_edge_max(const int* __restrict__ src, const int* __restrict__ dst,
                           int E, int Nn, int Hh, const float* __restrict__ es,
                           const float* __restrict__ ed, unsigned* __restrict__ menc) {
  long long i = (long long)blockIdx.x * blockDim.x + threadIdx.x;
  long long total = (long long)(E + Nn) * Hh;
  if (i >= total) return;
  int ei = (int)(i / Hh), hh = (int)(i % Hh);
  int s = (ei < E) ? src[ei] : (ei - E);
  int d = (ei < E) ? dst[ei] : (ei - E);
  float e = es[(long long)s * Hh + hh] + ed[(long long)d * Hh + hh];
  e = (e > 0.f) ? e : NEG_SLOPE * e;
  atomicMax(&menc[(long long)d * Hh + hh], enc_f32(e));
}

// pass 2: ex = exp(e - max); store per-edge; atomic-add denominator
__global__ void k_edge_expsum(const int* __restrict__ src, const int* __restrict__ dst,
                              int E, int Nn, int Hh, const float* __restrict__ es,
                              const float* __restrict__ ed, const unsigned* __restrict__ menc,
                              float* __restrict__ exb, float* __restrict__ denom) {
  long long i = (long long)blockIdx.x * blockDim.x + threadIdx.x;
  long long total = (long long)(E + Nn) * Hh;
  if (i >= total) return;
  int ei = (int)(i / Hh), hh = (int)(i % Hh);
  int s = (ei < E) ? src[ei] : (ei - E);
  int d = (ei < E) ? dst[ei] : (ei - E);
  float e = es[(long long)s * Hh + hh] + ed[(long long)d * Hh + hh];
  e = (e > 0.f) ? e : NEG_SLOPE * e;
  float m = dec_f32(menc[(long long)d * Hh + hh]);
  float ex = __expf(e - m);
  exb[i] = ex;
  atomicAdd(&denom[(long long)d * Hh + hh], ex);
}

// pass 3: acc[dst,h,:] += alpha * h[src,h,:]   (one wave per (edge,head))
__global__ void k_edge_scatter(const int* __restrict__ src, const int* __restrict__ dst,
                               int E, int Nn, int Hh, int Cc, const float* __restrict__ h,
                               const float* __restrict__ exb, const float* __restrict__ denom,
                               float* __restrict__ acc) {
  long long wave = (blockIdx.x * (long long)blockDim.x + threadIdx.x) >> 5;
  int lane = threadIdx.x & 31;
  long long total = (long long)(E + Nn) * Hh;
  if (wave >= total) return;
  int ei = (int)(wave / Hh), hh = (int)(wave % Hh);
  int s = (ei < E) ? src[ei] : (ei - E);
  int d = (ei < E) ? dst[ei] : (ei - E);
  float alpha = exb[wave] / (denom[(long long)d * Hh + hh] + 1e-16f);
  const float* hp = h   + ((long long)s * Hh + hh) * Cc;
  float*       ap = acc + ((long long)d * Hh + hh) * Cc;
  for (int c = lane; c < Cc; c += 32)
    atomicAdd(ap + c, alpha * hp[c]);
}

// mean over heads + bias; write f32 final output and/or bf16 input for next layer
__global__ void k_finalize(const float* __restrict__ acc, const float* __restrict__ bias,
                           float* __restrict__ outf, bf16_t* __restrict__ outb,
                           int Nn, int Hh, int Cc) {
  long long i = (long long)blockIdx.x * blockDim.x + threadIdx.x;
  long long total = (long long)Nn * Cc;
  if (i >= total) return;
  int n = (int)(i / Cc), c = (int)(i % Cc);
  float sum = 0.f;
  for (int hh = 0; hh < Hh; ++hh)
    sum += acc[((long long)n * Hh + hh) * Cc + c];
  float v = sum / (float)Hh + bias[c];
  if (outf) outf[i] = v;
  if (outb) outb[i] = (bf16_t)v;
}

static inline int gblocks(long long threads, int bt) {
  return (int)((threads + bt - 1) / bt);
}

extern "C" void kernel_launch(void* const* d_in, const int* in_sizes, int n_in,
                              void* d_out, int out_size, void* d_ws, size_t ws_size,
                              hipStream_t stream) {
  (void)n_in; (void)out_size; (void)ws_size;
  const float* x    = (const float*)d_in[0];
  const int*   eidx = (const int*)d_in[1];
  int E = in_sizes[1] / 2;
  const int* srcA = eidx;
  const int* dstA = eidx + E;

  int fi0 = in_sizes[2] / in_sizes[3];   // F_IN (W0 elems / as0 elems)
  int H   = in_sizes[3] / in_sizes[5];   // heads (as0 elems / b0 elems)
  int N   = in_sizes[0] / fi0;           // nodes

  struct Layer { const float *W, *as, *ad, *b; int K, C; } L[3];
  L[0] = { (const float*)d_in[2],  (const float*)d_in[3],  (const float*)d_in[4],
           (const float*)d_in[5],  fi0,                         in_sizes[5]  };
  L[1] = { (const float*)d_in[6],  (const float*)d_in[7],  (const float*)d_in[8],
           (const float*)d_in[9],  in_sizes[6]  / in_sizes[7],  in_sizes[9]  };
  L[2] = { (const float*)d_in[10], (const float*)d_in[11], (const float*)d_in[12],
           (const float*)d_in[13], in_sizes[10] / in_sizes[11], in_sizes[13] };

  int Kmax = L[0].K, Cmax = L[0].C;
  for (int l = 1; l < 3; ++l) { if (L[l].K > Kmax) Kmax = L[l].K; if (L[l].C > Cmax) Cmax = L[l].C; }
  int NoutMax = H * Cmax;

  // ---- workspace carve-out ----
  size_t off = 0; char* wsb = (char*)d_ws;
  auto alloc = [&](size_t bytes) -> void* {
    off = (off + 255) & ~(size_t)255;
    void* p = wsb + off; off += bytes; return p;
  };
  bf16_t*   xb    = (bf16_t*)alloc((size_t)N * Kmax * sizeof(bf16_t));       // GEMM A (bf16)
  bf16_t*   Wt    = (bf16_t*)alloc((size_t)NoutMax * Kmax * sizeof(bf16_t)); // GEMM B^T (bf16)
  float*    hbuf  = (float*)alloc((size_t)N * NoutMax * sizeof(float));      // h = x@W
  float*    acc   = (float*)alloc((size_t)N * NoutMax * sizeof(float));      // scatter accum
  float*    es    = (float*)alloc((size_t)N * H * sizeof(float));
  float*    ed    = (float*)alloc((size_t)N * H * sizeof(float));
  unsigned* menc  = (unsigned*)alloc((size_t)N * H * sizeof(unsigned));
  float*    denom = (float*)alloc((size_t)N * H * sizeof(float));
  float*    exb   = (float*)alloc((size_t)(E + N) * H * sizeof(float));

  const int BT = 256;

  // layer-0 input -> bf16
  k_f32_to_bf16<<<gblocks((long long)N * fi0, BT), BT, 0, stream>>>(x, xb, (long long)N * fi0);

  for (int l = 0; l < 3; ++l) {
    int K = L[l].K, C = L[l].C, Nout = H * C;

    k_transpose_w<<<gblocks((long long)K * Nout, BT), BT, 0, stream>>>(L[l].W, Wt, K, Nout);

    long long strips = (long long)((N + 15) / 16) * (Nout / 64);   // 16x64 per wave
    k_gemm_wmma<<<gblocks(strips * 32, BT), BT, 0, stream>>>(xb, Wt, hbuf, N, K, Nout);

    k_node_logits<<<gblocks((long long)N * H * 32, BT), BT, 0, stream>>>(
        hbuf, L[l].as, L[l].ad, es, ed, N, H, C);

    k_fill_u32<<<gblocks((long long)N * H, BT), BT, 0, stream>>>(menc, 0x007FFFFFu, (long long)N * H);
    k_fill_u32<<<gblocks((long long)N * H, BT), BT, 0, stream>>>((unsigned*)denom, 0u, (long long)N * H);
    k_fill_u32<<<gblocks((long long)N * Nout, BT), BT, 0, stream>>>((unsigned*)acc, 0u, (long long)N * Nout);

    long long EH = (long long)(E + N) * H;
    k_edge_max<<<gblocks(EH, BT), BT, 0, stream>>>(srcA, dstA, E, N, H, es, ed, menc);
    k_edge_expsum<<<gblocks(EH, BT), BT, 0, stream>>>(srcA, dstA, E, N, H, es, ed, menc, exb, denom);
    k_edge_scatter<<<gblocks(EH * 32, BT), BT, 0, stream>>>(srcA, dstA, E, N, H, C,
                                                            hbuf, exb, denom, acc);

    k_finalize<<<gblocks((long long)N * C, BT), BT, 0, stream>>>(
        acc, L[l].b,
        (l == 2) ? (float*)d_out : nullptr,   // final layer -> f32 output
        (l < 2) ? xb : nullptr,               // else -> bf16 input of next layer
        N, H, C);
  }
}
